// MPNNGNN_18889266168161
// MI455X (gfx1250) — matrix operand order repaired and checked by reading, not compiled, not run on hardware
//
#include <hip/hip_runtime.h>

typedef __attribute__((ext_vector_type(16))) _Float16 v16h;
typedef __attribute__((ext_vector_type(8)))  float    v8f;

#define NN   4096
#define NE   32768
#define DD   64
#define EHH  128

// ---------------------------------------------------------------------------
// A-fragment loader: 16x32 f16 tile (M x K) from LDS, contiguous K order (B packed to match):
//   lane 0-15  row M=lane,    halves i: K = i (0..15)
//   lane 16-31 row M=lane-16, halves i: K = 16 + i (16..31)
// ---------------------------------------------------------------------------
__device__ __forceinline__ v16h load_a_frag(const _Float16* base, int stride,
                                            int m0, int kbase, int lane) {
  const int lm = lane & 15;
  const int hi = lane >> 4;
  v16h a;
#pragma unroll
  for (int i = 0; i < 16; ++i) {
    int k = hi * 16 + i;
    a[i] = base[(m0 + lm) * stride + kbase + k];
  }
  return a;
}

// ---------------------------------------------------------------------------
// Pack a KxNcols fp32 weight matrix into WMMA B-fragment order (f16):
// fragment fr = ct*(K/32)+kk holds the 32x16 block (K rows kk*32..+32,
// cols ct*16..+16); lane L half i -> K = kk*32 + (L<16?0:16)+i, N = ct*16+(L&15).
// Stored so each lane's v16h is one contiguous 32-byte load.
// ---------------------------------------------------------------------------
__global__ void __launch_bounds__(256) pack_b_kernel(const float* __restrict__ W,
                                                     _Float16* __restrict__ P,
                                                     int K, int Ncols) {
  int idx = blockIdx.x * 256 + threadIdx.x;
  int total = K * Ncols;
  if (idx >= total) return;
  int i  = idx & 15;
  int L  = (idx >> 4) & 31;
  int fr = idx >> 9;
  int KC = K >> 5;
  int kk = fr % KC;
  int ct = fr / KC;
  int k = kk * 32 + ((L < 16) ? 0 : 16) + i;
  int n = ct * 16 + (L & 15);
  P[idx] = (_Float16)W[k * Ncols + n];
}

// ---------------------------------------------------------------------------
// h = relu(node_feats @ W_p + b_p)      (4096x64)@(64x64)
// ---------------------------------------------------------------------------
__global__ void __launch_bounds__(64) proj_node_kernel(const float* __restrict__ x,
                                                       const float* __restrict__ W,
                                                       const float* __restrict__ b,
                                                       float* __restrict__ h) {
  __shared__ float xs[DD];
  int n = blockIdx.x, f = threadIdx.x;
  xs[f] = x[n * DD + f];
  __syncthreads();
  float a = b[f];
#pragma unroll 8
  for (int k = 0; k < DD; ++k) a = fmaf(xs[k], W[k * DD + f], a);
  h[n * DD + f] = fmaxf(a, 0.f);
}

// ---------------------------------------------------------------------------
// e = relu(edge_attr @ W_pe + b_pe)     (32768x16)@(16x128)
// ---------------------------------------------------------------------------
__global__ void __launch_bounds__(128) proj_edge_kernel(const float* __restrict__ x,
                                                        const float* __restrict__ W,
                                                        const float* __restrict__ b,
                                                        float* __restrict__ e) {
  __shared__ float xs[16];
  int ed = blockIdx.x, f = threadIdx.x;
  if (f < 16) xs[f] = x[ed * 16 + f];
  __syncthreads();
  float a = b[f];
#pragma unroll
  for (int k = 0; k < 16; ++k) a = fmaf(xs[k], W[k * EHH + f], a);
  e[ed * EHH + f] = fmaxf(a, 0.f);
}

// ---------------------------------------------------------------------------
// Fused NNConv message kernel.
// Block = 32 edges, 256 threads (8 waves). Waves 0-3 -> edges 0-15,
// waves 4-7 -> edges 16-31; (wave&3) selects 16-wide f column tile.
// For each d (0..63): W_chunk(16x16) = relu(e_tile(16x128) @ Wnn[:, d*64+f..]
// + b) via 4 chained v_wmma_f32_16x16x32_f16, then msg += h[row][d]*W_chunk.
// Scatter-add msg into agg[col] with global float atomics.
// ---------------------------------------------------------------------------
#define ELS 136   // padded LDS row stride (halves) to avoid bank conflicts
__global__ void __launch_bounds__(256) msg_kernel(const float* __restrict__ e,
                                                  const float* __restrict__ h,
                                                  const int* __restrict__ rowI,
                                                  const int* __restrict__ colI,
                                                  const _Float16* __restrict__ WnnP,
                                                  const float* __restrict__ b_nn,
                                                  float* __restrict__ agg) {
  __shared__ _Float16 e_lds[32 * ELS];
  __shared__ float    h_lds[32 * DD];

  const int tid   = threadIdx.x;
  const int lane  = tid & 31;
  const int wave  = tid >> 5;
  const int edge0 = blockIdx.x * 32;

  for (int idx = tid; idx < 32 * EHH; idx += 256) {
    int er = idx >> 7, c = idx & 127;
    e_lds[er * ELS + c] = (_Float16)e[(edge0 + er) * EHH + c];
  }
  for (int idx = tid; idx < 32 * DD; idx += 256) {
    int er = idx >> 6, c = idx & 63;
    h_lds[er * DD + c] = h[rowI[edge0 + er] * DD + c];
  }
  __syncthreads();

  const int esub  = (wave >> 2) * 16;   // which 16-edge half-tile
  const int fbase = (wave & 3) * 16;    // which 16-wide f tile
  const int lm    = lane & 15;
  const int hi    = lane >> 4;

  // A fragments are invariant across the d loop: e_tile 16x128 -> 4 K-chunks
  v16h afr[4];
#pragma unroll
  for (int kk = 0; kk < 4; ++kk)
    afr[kk] = load_a_frag(e_lds, ELS, esub, kk * 32, lane);

  const v16h* Bp = (const v16h*)WnnP;
  float msg[8];
#pragma unroll
  for (int j = 0; j < 8; ++j) msg[j] = 0.f;

  for (int d = 0; d < DD; ++d) {
    float bias = b_nn[d * DD + fbase + lm];   // column bias, same for all M
    v8f acc;
#pragma unroll
    for (int j = 0; j < 8; ++j) acc[j] = bias;
    int ct = d * 4 + (wave & 3);              // column tile into 4096-wide Wnn
#pragma unroll
    for (int kk = 0; kk < 4; ++kk) {
      v16h bfr = Bp[(ct * 4 + kk) * 32 + lane];
      acc = __builtin_amdgcn_wmma_f32_16x16x32_f16(false, afr[kk], false, bfr,
                                                   (short)0, acc, false, false);
    }
#pragma unroll
    for (int j = 0; j < 8; ++j) {
      int M = j + (hi ? 8 : 0);
      float w = fmaxf(acc[j], 0.f);
      msg[j] = fmaf(h_lds[(esub + M) * DD + d], w, msg[j]);
    }
  }

#pragma unroll
  for (int j = 0; j < 8; ++j) {
    int M = j + (hi ? 8 : 0);
    int tgt = colI[edge0 + esub + M];
    atomicAdd(&agg[tgt * DD + fbase + lm], msg[j]);
  }
}

// ---------------------------------------------------------------------------
// m = relu(agg + h @ W_root + b_conv); GRU cell (torch gate order r,z,n);
// h <- hidden_new (row-local, safe in place). 64 threads = one node.
// ---------------------------------------------------------------------------
__global__ void __launch_bounds__(64) node_gru_kernel(float* __restrict__ h,
                                                      const float* __restrict__ agg,
                                                      const float* __restrict__ W_root,
                                                      const float* __restrict__ b_conv,
                                                      const float* __restrict__ w_ih,
                                                      const float* __restrict__ w_hh,
                                                      const float* __restrict__ b_ih,
                                                      const float* __restrict__ b_hh) {
  __shared__ float hs[DD], ms[DD];
  int n = blockIdx.x, f = threadIdx.x;
  hs[f] = h[n * DD + f];
  __syncthreads();

  float a = agg[n * DD + f] + b_conv[f];
#pragma unroll 8
  for (int k = 0; k < DD; ++k) a = fmaf(hs[k], W_root[k * DD + f], a);
  ms[f] = fmaxf(a, 0.f);
  __syncthreads();

  float ir = b_ih[f], iz = b_ih[DD + f], in_ = b_ih[2 * DD + f];
  float hr = b_hh[f], hz = b_hh[DD + f], hn  = b_hh[2 * DD + f];
#pragma unroll 4
  for (int k = 0; k < DD; ++k) {
    float mk = ms[k], hk = hs[k];
    const float* wi = w_ih + k * 3 * DD;
    const float* wh = w_hh + k * 3 * DD;
    ir = fmaf(mk, wi[f], ir);       iz = fmaf(mk, wi[DD + f], iz);
    in_ = fmaf(mk, wi[2 * DD + f], in_);
    hr = fmaf(hk, wh[f], hr);       hz = fmaf(hk, wh[DD + f], hz);
    hn = fmaf(hk, wh[2 * DD + f], hn);
  }
  float r  = 1.f / (1.f + expf(-(ir + hr)));
  float z  = 1.f / (1.f + expf(-(iz + hz)));
  float nc = tanhf(in_ + r * hn);
  h[n * DD + f] = (1.f - z) * nc + z * hs[f];
}

// ---------------------------------------------------------------------------
// Edge update: e_new = relu(relu(cat(h[row],h[col],e) @ W_eu1 + b1) @ W_eu2 + b2)
// Block = 16 edges, 256 threads (8 waves); wave w owns output column tile w*16.
// Two WMMA GEMM phases with t1 staged in LDS (f16).
// ---------------------------------------------------------------------------
#define ALS 264   // padded stride for 256-wide cat tile
#define TLS 136   // padded stride for 128-wide t1 tile
__global__ void __launch_bounds__(256) edge_update_kernel(const float* __restrict__ h,
                                                          const float* __restrict__ e,
                                                          const int* __restrict__ rowI,
                                                          const int* __restrict__ colI,
                                                          const _Float16* __restrict__ Weu1P,
                                                          const float* __restrict__ b1,
                                                          const _Float16* __restrict__ Weu2P,
                                                          const float* __restrict__ b2,
                                                          float* __restrict__ e_out) {
  __shared__ _Float16 a_lds[16 * ALS];
  __shared__ _Float16 t_lds[16 * TLS];

  const int tid   = threadIdx.x;
  const int lane  = tid & 31;
  const int wave  = tid >> 5;
  const int edge0 = blockIdx.x * 16;

  for (int idx = tid; idx < 16 * 256; idx += 256) {
    int er = idx >> 8, c = idx & 255;
    int ge = edge0 + er;
    float v;
    if (c < DD)            v = h[rowI[ge] * DD + c];
    else if (c < 2 * DD)   v = h[colI[ge] * DD + (c - DD)];
    else                   v = e[ge * EHH + (c - 2 * DD)];
    a_lds[er * ALS + c] = (_Float16)v;
  }
  __syncthreads();

  const int fbase = wave * 16;
  const int lm    = lane & 15;
  const int hi    = lane >> 4;
  const v16h* Bp1 = (const v16h*)Weu1P;
  const v16h* Bp2 = (const v16h*)Weu2P;

  // phase 1: t1 = relu(A(16x256) @ W_eu1 + b1), this wave's 16 columns
  {
    float bias = b1[fbase + lm];
    v8f acc;
#pragma unroll
    for (int j = 0; j < 8; ++j) acc[j] = bias;
#pragma unroll
    for (int kk = 0; kk < 8; ++kk) {
      v16h afr = load_a_frag(a_lds, ALS, 0, kk * 32, lane);
      v16h bfr = Bp1[(wave * 8 + kk) * 32 + lane];
      acc = __builtin_amdgcn_wmma_f32_16x16x32_f16(false, afr, false, bfr,
                                                   (short)0, acc, false, false);
    }
#pragma unroll
    for (int j = 0; j < 8; ++j) {
      int M = j + (hi ? 8 : 0);
      t_lds[M * TLS + fbase + lm] = (_Float16)fmaxf(acc[j], 0.f);
    }
  }
  __syncthreads();

  // phase 2: e_new = relu(t1(16x128) @ W_eu2 + b2)
  {
    float bias = b2[fbase + lm];
    v8f acc;
#pragma unroll
    for (int j = 0; j < 8; ++j) acc[j] = bias;
#pragma unroll
    for (int kk = 0; kk < 4; ++kk) {
      v16h afr = load_a_frag(t_lds, TLS, 0, kk * 32, lane);
      v16h bfr = Bp2[(wave * 4 + kk) * 32 + lane];
      acc = __builtin_amdgcn_wmma_f32_16x16x32_f16(false, afr, false, bfr,
                                                   (short)0, acc, false, false);
    }
#pragma unroll
    for (int j = 0; j < 8; ++j) {
      int M = j + (hi ? 8 : 0);
      e_out[(edge0 + M) * EHH + fbase + lm] = fmaxf(acc[j], 0.f);
    }
  }
}

// ---------------------------------------------------------------------------
extern "C" void kernel_launch(void* const* d_in, const int* in_sizes, int n_in,
                              void* d_out, int out_size, void* d_ws, size_t ws_size,
                              hipStream_t stream) {
  (void)in_sizes; (void)n_in; (void)out_size; (void)ws_size;

  const float* node_feats = (const float*)d_in[0];
  const float* edge_attr  = (const float*)d_in[1];
  const int*   edge_index = (const int*)d_in[2];
  const float* W_p    = (const float*)d_in[3];
  const float* b_p    = (const float*)d_in[4];
  const float* W_pe   = (const float*)d_in[5];
  const float* b_pe   = (const float*)d_in[6];
  const float* W_nn   = (const float*)d_in[7];
  const float* b_nn   = (const float*)d_in[8];
  const float* W_root = (const float*)d_in[9];
  const float* b_conv = (const float*)d_in[10];
  const float* w_ih   = (const float*)d_in[11];
  const float* w_hh   = (const float*)d_in[12];
  const float* b_ih   = (const float*)d_in[13];
  const float* b_hh   = (const float*)d_in[14];
  const float* W_eu1  = (const float*)d_in[15];
  const float* b_eu1  = (const float*)d_in[16];
  const float* W_eu2  = (const float*)d_in[17];
  const float* b_eu2  = (const float*)d_in[18];

  const int* rowI = edge_index;        // sources
  const int* colI = edge_index + NE;   // targets

  char* ws = (char*)d_ws;
  size_t off = 0;
  auto alloc = [&](size_t bytes) -> void* {
    void* p = ws + off;
    off = (off + bytes + 255) & ~(size_t)255;
    return p;
  };
  float*    h     = (float*)alloc((size_t)NN * DD * 4);
  float*    agg   = (float*)alloc((size_t)NN * DD * 4);
  float*    e_a   = (float*)alloc((size_t)NE * EHH * 4);
  float*    e_b   = (float*)alloc((size_t)NE * EHH * 4);
  _Float16* WnnP  = (_Float16*)alloc((size_t)128 * 4096 * 2);
  _Float16* Weu1P = (_Float16*)alloc((size_t)256 * 128 * 2);
  _Float16* Weu2P = (_Float16*)alloc((size_t)128 * 128 * 2);

  // pack weights into WMMA B-fragment order (f16)
  pack_b_kernel<<<(128 * 4096 + 255) / 256, 256, 0, stream>>>(W_nn,  WnnP,  128, 4096);
  pack_b_kernel<<<(256 * 128  + 255) / 256, 256, 0, stream>>>(W_eu1, Weu1P, 256, 128);
  pack_b_kernel<<<(128 * 128  + 255) / 256, 256, 0, stream>>>(W_eu2, Weu2P, 128, 128);

  // input projections
  proj_node_kernel<<<NN, 64, 0, stream>>>(node_feats, W_p, b_p, h);
  proj_edge_kernel<<<NE, 128, 0, stream>>>(edge_attr, W_pe, b_pe, e_a);

  float* e_cur = e_a;
  float* e_nxt = e_b;
  for (int step = 0; step < 3; ++step) {
    hipMemsetAsync(agg, 0, (size_t)NN * DD * 4, stream);
    msg_kernel<<<NE / 32, 256, 0, stream>>>(e_cur, h, rowI, colI, WnnP, b_nn, agg);
    node_gru_kernel<<<NN, 64, 0, stream>>>(h, agg, W_root, b_conv,
                                           w_ih, w_hh, b_ih, b_hh);
    edge_update_kernel<<<NE / 16, 256, 0, stream>>>(h, e_cur, rowI, colI,
                                                    Weu1P, b_eu1, Weu2P, b_eu2, e_nxt);
    float* t = e_cur; e_cur = e_nxt; e_nxt = t;
  }

  hipMemcpyAsync(d_out, h, (size_t)NN * DD * 4, hipMemcpyDeviceToDevice, stream);
  hipMemcpyAsync((float*)d_out + (size_t)NN * DD, e_cur,
                 (size_t)NE * EHH * 4, hipMemcpyDeviceToDevice, stream);
}
